// RecurrentActorCritic_37778532335858
// MI455X (gfx1250) — matrix operand order repaired and checked by reading, not compiled, not run on hardware
//
#include <hip/hip_runtime.h>
#include <stdint.h>

// ---------------- problem constants ----------------
#define kB 128
#define kT 1024
#define kD 256
#define kH 256
#define kG 1024            // 4*H gate width
#define kK 512             // D + H stacked contraction
#define kA 8
#define kW 256
#define NWG 8              // workgroups per LSTM
#define HU_PER_WG 32       // hidden units per WG (=> 128 z-columns: 4 gates x 32)
#define A_ROW 528          // bytes per LDS A row: 16 ktiles * 32B + 16B pad (bank decorrelation)
#define WLDS_ENTRIES (16 * 8 * 32 * 16)               // swizzled B operand slice, bf16
#define SMEM_BYTES (WLDS_ENTRIES * 2 + kB * A_ROW)    // 131072 + 67584 = 198656 B (< 320KB WGP LDS)

// ---------------- d_out layout (flat f32, tuple order) ----------------
#define OFF_MEAN 0
#define OFF_LOGSTD 1024
#define OFF_VALUE 1032
#define OFF_ACTOR_C 1160
#define OFF_ACTOR_H (OFF_ACTOR_C + kB * kH)
#define OFF_CRITIC_C (OFF_ACTOR_H + kB * kH)
#define OFF_CRITIC_H (OFF_CRITIC_C + kB * kH)

// ---------------- workspace layout (bytes) ----------------
#define WS_WCOMB 0                                   // 2 LSTMs * [512][1024] bf16 = 2 MiB
#define WS_HGLOB (2 * kK * kG * 2)
#define WS_CTR (WS_HGLOB + 2 * 2 * kB * kH * 2)      // + 2*2*[128][256] bf16 = 256 KiB

typedef __bf16 v16bf __attribute__((ext_vector_type(16)));
typedef float v8f __attribute__((ext_vector_type(8)));
typedef int v4i __attribute__((vector_size(16)));    // matches async-LDS builtin pointee

#define GAS __attribute__((address_space(1)))
#define LAS __attribute__((address_space(3)))

// gfx1250 async global->LDS copy path (ASYNCcnt-tracked), guarded so the file
// still compiles on toolchains that lack the builtins.
#if __has_builtin(__builtin_amdgcn_global_load_async_to_lds_b128) && \
    __has_builtin(__builtin_amdgcn_s_wait_asynccnt)
#define HAVE_ASYNC_LDS 1
#else
#define HAVE_ASYNC_LDS 0
#endif

__device__ __forceinline__ unsigned short f2bf(float f) {
  unsigned u = __float_as_uint(f);
  u += 0x7FFFu + ((u >> 16) & 1u);     // round-to-nearest-even
  return (unsigned short)(u >> 16);
}

// A-operand LDS address for logical element (row m, contraction index K).
// Layout per row: [ktile(16)][half(2)][16 bf16] so each lane's v16bf fetch is one
// contiguous 32B chunk matching the 16-bit A VGPR layout:
//   e<8  -> K = kt*32 + 8*half + e
//   e>=8 -> K = kt*32 + 16 + 8*half + (e-8)
// Any K-run [8a, 8a+8) is 16B-contiguous in both global row-major and this layout.
__device__ __forceinline__ int aOff(int m, int K) {
  int kt = K >> 5;
  int r = K & 31;
  int half = (r >> 3) & 1;
  int e = (r & 7) | ((r >> 4) << 3);
  return m * A_ROW + ((kt * 2 + half) << 5) + (e << 1);
}

// ---------------- kernel: zero the barrier counters ----------------
__global__ void ctr_init(unsigned* __restrict__ ctr) {
  if (threadIdx.x < 32) ctr[threadIdx.x] = 0u;
}

// ---------------- kernel: build combined bf16 weights [Wx ; Wh] per LSTM ----------------
__global__ void wcomb_build(const float* __restrict__ Wx_a, const float* __restrict__ Wh_a,
                            const float* __restrict__ Wx_c, const float* __restrict__ Wh_c,
                            unsigned short* __restrict__ wcomb) {
  int id = blockIdx.x * blockDim.x + threadIdx.x;   // over 2*512*1024
  int n = id & (kG - 1);
  int k = (id >> 10) & (kK - 1);
  int l = id >> 19;
  const float* Wx = l ? Wx_c : Wx_a;
  const float* Wh = l ? Wh_c : Wh_a;
  float v = (k < kD) ? Wx[k * kG + n] : Wh[(k - kD) * kG + n];
  wcomb[id] = f2bf(v);
}

// Producer/consumer step barrier among the NWG workgroups of one LSTM.
// If dispatched as a workgroup cluster (8 WGs/LSTM on one SE), use the HW
// cluster barrier; on a plain dispatch (cluster_id==0) fall back to an
// agent-scope atomic counter with s_sleep backoff.
__device__ __forceinline__ void lstm_barrier(unsigned* ctr, unsigned target) {
  __threadfence();          // release h writes to device scope
  __syncthreads();
  if (threadIdx.x == 0) {
    if (__builtin_amdgcn_cluster_id_x() != 0) {
      __builtin_amdgcn_s_cluster_barrier();  // s_barrier_signal -3 ; s_barrier_wait -3
      __hip_atomic_fetch_add(ctr, 1u, __ATOMIC_RELAXED, __HIP_MEMORY_SCOPE_AGENT);
    } else {
      __hip_atomic_fetch_add(ctr, 1u, __ATOMIC_RELEASE, __HIP_MEMORY_SCOPE_AGENT);
      while (__hip_atomic_load(ctr, __ATOMIC_ACQUIRE, __HIP_MEMORY_SCOPE_AGENT) < target)
        __builtin_amdgcn_s_sleep(2);
    }
  }
  __syncthreads();
  __threadfence();          // acquire peers' h writes
}

// ---------------- kernel: fused dual-LSTM scan ----------------
// grid.x = 16: blocks [0,8) actor, [8,16) critic. Each WG owns 32 hidden units.
// Per step: gates z = [x_t ; h] @ [Wx;Wh] + b via v_wmma_f32_16x16x32_bf16,
// weights resident in LDS (pre-swizzled B layout), cell state c in registers.
__global__ __launch_bounds__(256, 1) void lstm_scan(
    const float* __restrict__ x,
    const float* __restrict__ c0_a, const float* __restrict__ h0_a,
    const float* __restrict__ c0_c, const float* __restrict__ h0_c,
    const float* __restrict__ b_a, const float* __restrict__ b_c,
    const unsigned short* __restrict__ wcomb,
    unsigned short* __restrict__ hglob,
    unsigned* __restrict__ ctrs,
    float* __restrict__ out) {
  extern __shared__ char smem[];
  unsigned short* wlds = (unsigned short*)smem;          // 128KB swizzled weight slice
  char* aB = smem + WLDS_ENTRIES * 2;                    // 66KB A operand staging

  const int tid = threadIdx.x;
  const int l = blockIdx.x / NWG;
  const int sl = blockIdx.x % NWG;
  const int huB = sl * HU_PER_WG;
  const int wv = tid >> 5;        // wave id = M-tile
  const int lane = tid & 31;
  const int half = lane >> 4;
  const int nlo = lane & 15;

  const float* c0 = l ? c0_c : c0_a;
  const float* h0 = l ? h0_c : h0_a;
  const float* bias = l ? b_c : b_a;
  const unsigned short* wsrc = wcomb + (size_t)l * kK * kG;
  unsigned short* hg = hglob + (size_t)l * 2 * kB * kH;  // double-buffered h (bf16)
  unsigned* ctr = ctrs + l * 16;
  float* cOut = out + (l ? OFF_CRITIC_C : OFF_ACTOR_C);
  float* hOut = out + (l ? OFF_CRITIC_H : OFF_ACTOR_H);

  // Load + swizzle this WG's weight slice into exact WMMA B-operand layout:
  // entry (kt, nt, lane, e) = W[kt*32 + 8*(lane/16) + (e&7) + 16*(e/8)][gate*256 + huB + sub*16 + lane%16]
  // (gathered: element stride in global is 2KB, so this stays on the VALU path;
  // it runs once per launch.)
  for (int i = tid; i < WLDS_ENTRIES; i += 256) {
    int e = i & 15;
    int ln = (i >> 4) & 31;
    int nt = (i >> 9) & 7;
    int kt = i >> 12;
    int k = kt * 32 + ((ln >> 4) << 3) + (e & 7) + ((e >> 3) << 4);
    int n = (nt >> 1) * kH + huB + (nt & 1) * 16 + (ln & 15);
    wlds[i] = wsrc[k * kG + n];
  }

  // Cell state in registers: thread (wv,lane) owns (m = wv*16 + r + 8*half,
  // hu = huB + t*16 + nlo) for t in [0,2), r in [0,8)  -- matches the f32 C/D layout.
  float cReg[2][8];
#pragma unroll
  for (int t = 0; t < 2; ++t)
#pragma unroll
    for (int r = 0; r < 8; ++r) {
      int m = wv * 16 + r + 8 * half;
      int hu = huB + t * 16 + nlo;
      cReg[t][r] = c0[m * kH + hu];
    }
  // Publish h0 slice (bf16) into buffer 0.
  for (int i = tid; i < kB * HU_PER_WG; i += 256) {
    int m = i >> 5, j = i & 31;
    hg[m * kH + huB + j] = f2bf(h0[m * kH + huB + j]);
  }
  // Per-N-tile bias value (same for every row of a C tile column).
  float bv[8];
#pragma unroll
  for (int nt = 0; nt < 8; ++nt)
    bv[nt] = bias[(nt >> 1) * kH + huB + (nt & 1) * 16 + nlo];

  lstm_barrier(ctr, NWG);   // all h0 slices visible

  for (int s = 0; s < kT; ++s) {
    const unsigned short* hread = hg + (s & 1) * (kB * kH);
    unsigned short* hwrite = hg + ((s + 1) & 1) * (kB * kH);

    // Stage h -> A[:, 256:512): pure copy, so use the ASYNCcnt-tracked
    // global->LDS DMA path (global_load_async_to_lds_b128) when available.
#if HAVE_ASYNC_LDS
    for (int i = tid; i < kB * (kH / 8); i += 256) {
      int m = i >> 5, c8 = (i & 31) << 3;      // 8 consecutive K = 16B in both layouts
      const unsigned short* src = hread + m * kH + c8;
      char* dst = aB + aOff(m, kD + c8);
      __builtin_amdgcn_global_load_async_to_lds_b128(
          (GAS v4i*)(uintptr_t)src, (LAS v4i*)(unsigned)(uintptr_t)dst, 0, 0);
    }
#else
    for (int i = tid; i < kB * (kH / 2); i += 256) {
      int m = i >> 7, c2 = (i & 127) << 1;
      unsigned v = *(const unsigned*)(hread + m * kH + c2);
      *(unsigned*)(aB + aOff(m, kD + c2)) = v;
    }
#endif
    // Stage x_t (f32 -> bf16 conversion in flight) -> A[:, 0:256).
    for (int i = tid; i < kB * (kD / 4); i += 256) {
      int m = i >> 6, d4 = (i & 63) << 2;
      float4 v = *(const float4*)(x + ((size_t)m * kT + s) * kD + d4);
      unsigned p0 = (unsigned)f2bf(v.x) | ((unsigned)f2bf(v.y) << 16);
      unsigned p1 = (unsigned)f2bf(v.z) | ((unsigned)f2bf(v.w) << 16);
      *(uint2*)(aB + aOff(m, d4)) = make_uint2(p0, p1);
    }
    if (tid < kB && s + 1 < kT)
      __builtin_prefetch(x + ((size_t)tid * kT + (s + 1)) * kD, 0, 1);  // global_prefetch_b8
#if HAVE_ASYNC_LDS
    __builtin_amdgcn_s_wait_asynccnt(0);   // async LDS writes visible before barrier
#endif
    __syncthreads();

    // z = A @ Wslice + bias : 8 N-tiles x 16 k-steps of 16x16x32 bf16 WMMA.
    v8f acc[8];
#pragma unroll
    for (int nt = 0; nt < 8; ++nt) {
      v8f c;
#pragma unroll
      for (int j = 0; j < 8; ++j) c[j] = bv[nt];
      acc[nt] = c;
    }
    const int mrow = wv * 16 + nlo;
#pragma unroll 4
    for (int kt = 0; kt < 16; ++kt) {
      v16bf a = *(const v16bf*)(aB + mrow * A_ROW + ((kt * 2 + half) << 5));
#pragma unroll
      for (int nt = 0; nt < 8; ++nt) {
        v16bf b = *(const v16bf*)(wlds + (((kt * 8 + nt) * 32 + lane) << 4));
        acc[nt] = __builtin_amdgcn_wmma_f32_16x16x32_bf16(
            false, a, false, b, (short)0, acc[nt], false, false);
      }
    }

    // Gate nonlinearities + state update, directly on the C layout.
#pragma unroll
    for (int t = 0; t < 2; ++t) {
#pragma unroll
      for (int r = 0; r < 8; ++r) {
        float zi = acc[0 + t][r];
        float zf = acc[2 + t][r];
        float zg = acc[4 + t][r];
        float zo = acc[6 + t][r];
        float si = 1.f / (1.f + __expf(-zi));
        float sf = 1.f / (1.f + __expf(-zf));
        float so = 1.f / (1.f + __expf(-zo));
        float cN = sf * cReg[t][r] + si * tanhf(zg);
        float hN = so * tanhf(cN);
        cReg[t][r] = cN;
        int m = wv * 16 + r + 8 * half;
        int hu = huB + t * 16 + nlo;
        hwrite[m * kH + hu] = f2bf(hN);
        if (s == kT - 1) {
          cOut[m * kH + hu] = cN;
          hOut[m * kH + hu] = hN;
        }
      }
    }
    lstm_barrier(ctr, (unsigned)(NWG * (s + 2)));
  }
}

// ---------------- kernel: tanh-MLP heads + log_std passthrough ----------------
__global__ __launch_bounds__(256) void heads_kernel(
    const float* __restrict__ Wm1, const float* __restrict__ bm1,
    const float* __restrict__ Wm2, const float* __restrict__ bm2,
    const float* __restrict__ Wv1, const float* __restrict__ bv1,
    const float* __restrict__ Wv2, const float* __restrict__ bv2,
    const float* __restrict__ log_std, float* __restrict__ out) {
  __shared__ float ta[kW], tc[kW];
  const int b = blockIdx.x, w = threadIdx.x;
  const float* hA = out + OFF_ACTOR_H + b * kH;
  const float* hC = out + OFF_CRITIC_H + b * kH;
  float sa = bm1[w], sc = bv1[w];
  for (int d = 0; d < kH; ++d) {
    sa = fmaf(hA[d], Wm1[d * kW + w], sa);
    sc = fmaf(hC[d], Wv1[d * kW + w], sc);
  }
  ta[w] = tanhf(sa);
  tc[w] = tanhf(sc);
  __syncthreads();
  if (w < kA) {
    float m = bm2[w];
    for (int j = 0; j < kW; ++j) m = fmaf(ta[j], Wm2[j * kA + w], m);
    out[OFF_MEAN + b * kA + w] = m;
  } else if (w == kA) {
    float v = bv2[0];
    for (int j = 0; j < kW; ++j) v = fmaf(tc[j], Wv2[j], v);
    out[OFF_VALUE + b] = v;
  }
  if (b == 0 && w >= 16 && w < 16 + kA) out[OFF_LOGSTD + (w - 16)] = log_std[w - 16];
}

extern "C" void kernel_launch(void* const* d_in, const int* in_sizes, int n_in,
                              void* d_out, int out_size, void* d_ws, size_t ws_size,
                              hipStream_t stream) {
  (void)in_sizes; (void)n_in; (void)out_size; (void)ws_size;
  const float* x = (const float*)d_in[0];
  const float* a_c0 = (const float*)d_in[1];
  const float* a_h0 = (const float*)d_in[2];
  const float* c_c0 = (const float*)d_in[3];
  const float* c_h0 = (const float*)d_in[4];
  const float* Wx_a = (const float*)d_in[5];
  const float* Wh_a = (const float*)d_in[6];
  const float* b_a = (const float*)d_in[7];
  const float* Wx_c = (const float*)d_in[8];
  const float* Wh_c = (const float*)d_in[9];
  const float* b_c = (const float*)d_in[10];
  const float* Wm1 = (const float*)d_in[11];
  const float* bm1 = (const float*)d_in[12];
  const float* Wm2 = (const float*)d_in[13];
  const float* bm2 = (const float*)d_in[14];
  const float* Wv1 = (const float*)d_in[15];
  const float* bv1 = (const float*)d_in[16];
  const float* Wv2 = (const float*)d_in[17];
  const float* bv2 = (const float*)d_in[18];
  const float* lstd = (const float*)d_in[19];

  char* ws = (char*)d_ws;
  unsigned short* wcomb = (unsigned short*)(ws + WS_WCOMB);
  unsigned short* hglob = (unsigned short*)(ws + WS_HGLOB);
  unsigned* ctrs = (unsigned*)(ws + WS_CTR);
  float* out = (float*)d_out;

  // 194KB dynamic LDS per WG (gfx1250 WGP has 320KB) -- opt in every call
  // (idempotent, deterministic; not a stream op so graph-capture safe).
  (void)hipFuncSetAttribute(reinterpret_cast<const void*>(lstm_scan),
                            hipFuncAttributeMaxDynamicSharedMemorySize, SMEM_BYTES);

  ctr_init<<<1, 32, 0, stream>>>(ctrs);
  wcomb_build<<<(2 * kK * kG) / 256, 256, 0, stream>>>(Wx_a, Wh_a, Wx_c, Wh_c, wcomb);
  lstm_scan<<<2 * NWG, 256, SMEM_BYTES, stream>>>(x, a_c0, a_h0, c_c0, c_h0, b_a, b_c,
                                                  wcomb, hglob, ctrs, out);
  heads_kernel<<<kB, 256, 0, stream>>>(Wm1, bm1, Wm2, bm2, Wv1, bv1, Wv2, bv2, lstd, out);
}
